// MoeLoraLayer_10831907521049
// MI455X (gfx1250) — matrix operand order repaired and checked by reading, not compiled, not run on hardware
//
#include <hip/hip_runtime.h>

typedef __bf16  v8bf  __attribute__((ext_vector_type(8)));
typedef __bf16  v16bf __attribute__((ext_vector_type(16)));
typedef float   v8f   __attribute__((ext_vector_type(8)));

#define T_TOK   16384          // B*S tokens
#define DDIM    3072
#define ODIM    3072
#define EEXP    8
#define RRANK   32
#define KLORA   (EEXP*RRANK)   // 256
#define K2      (DDIM + KLORA) // 3328
#define SCALING 2.0f

#define BM    128
#define BN    256
#define BK    32
#define LDSS  40   // BK + 8 pad; row stride 80B = 5*16B -> all frag loads stay 16B aligned

static __device__ __forceinline__ unsigned short f2bf(float f) {
    unsigned int u = __builtin_bit_cast(unsigned int, f);
    u += 0x7fffu + ((u >> 16) & 1u);      // round-to-nearest-even
    return (unsigned short)(u >> 16);
}

// LDS byte offset of a __shared__ pointer (generic LDS addresses carry the
// wave-relative LDS byte address in the low 32 bits).
static __device__ __forceinline__ unsigned lds_off(const void* p) {
    return (unsigned)(unsigned long long)(uintptr_t)p;
}

// CDNA5 async memory->LDS copy, tracked with ASYNCcnt (no VGPR staging).
static __device__ __forceinline__ void async_b128(unsigned lds, const unsigned short* g) {
    asm volatile("global_load_async_to_lds_b128 %0, %1, off"
                 :: "v"(lds), "v"(g) : "memory");
}
static __device__ __forceinline__ void wait_async0() {
    asm volatile("s_wait_asynccnt 0x0" ::: "memory");
}

// A fragment (16x32 bf16): lane<16 -> row=lane, K {0..7,16..23}; lane>=16 -> row=lane-16, K {8..15,24..31}
static __device__ __forceinline__ v16bf frag_a(const unsigned short* tile, int stride) {
    const int lane = threadIdx.x & 31;
    const unsigned short* p = tile + (lane & 15) * stride + ((lane >> 4) << 3);
    v8bf lo = *(const v8bf*)p;
    v8bf hi = *(const v8bf*)(p + 16);
    return __builtin_shufflevector(lo, hi, 0,1,2,3,4,5,6,7,8,9,10,11,12,13,14,15);
}

// B fragment (32x16 bf16 from row-major B^T=[N][K]): lane<16 -> col=lane, K {0..7,16..23}; lane>=16 -> K {8..15,24..31}
static __device__ __forceinline__ v16bf frag_b(const unsigned short* tile, int stride) {
    const int lane = threadIdx.x & 31;
    const unsigned short* p = tile + (lane & 15) * stride + ((lane >> 4) << 3);
    v8bf lo = *(const v8bf*)p;
    v8bf hi = *(const v8bf*)(p + 16);
    return __builtin_shufflevector(lo, hi, 0,1,2,3,4,5,6,7,8,9,10,11,12,13,14,15);
}

// ---------------------------------------------------------------------------
// Kernel 1: per-token router (softmax -> top2 -> renorm) + x fp32 -> bf16 into A'
// ---------------------------------------------------------------------------
__global__ __launch_bounds__(256) void route_convert_x(const float* __restrict__ x,
                                                       const float* __restrict__ gate_w,
                                                       unsigned short* __restrict__ Ap,
                                                       float* __restrict__ wte) {
    const int t   = blockIdx.x;
    const int tid = threadIdx.x;
    const float* xr = x + (size_t)t * DDIM;
    unsigned short* ar = Ap + (size_t)t * K2;

    float acc[EEXP];
#pragma unroll
    for (int e = 0; e < EEXP; ++e) acc[e] = 0.f;

#pragma unroll
    for (int i = 0; i < DDIM / 256; ++i) {
        const int d = tid + (i << 8);
        const float xv = xr[d];
        ar[d] = f2bf(xv);
#pragma unroll
        for (int e = 0; e < EEXP; ++e) acc[e] += xv * gate_w[e * DDIM + d];
    }

    __shared__ float red[EEXP * 256];
#pragma unroll
    for (int e = 0; e < EEXP; ++e) red[e * 256 + tid] = acc[e];
    __syncthreads();
    for (int s = 128; s > 0; s >>= 1) {
        if (tid < s) {
#pragma unroll
            for (int e = 0; e < EEXP; ++e) red[e * 256 + tid] += red[e * 256 + tid + s];
        }
        __syncthreads();
    }
    if (tid == 0) {
        float l[EEXP], p[EEXP];
        float m = red[0];
#pragma unroll
        for (int e = 0; e < EEXP; ++e) { l[e] = red[e * 256]; m = fmaxf(m, l[e]); }
#pragma unroll
        for (int e = 0; e < EEXP; ++e) p[e] = __expf(l[e] - m);
        int i1 = 0;
#pragma unroll
        for (int e = 1; e < EEXP; ++e) if (p[e] > p[i1]) i1 = e;
        int i2 = (i1 == 0) ? 1 : 0;
#pragma unroll
        for (int e = 0; e < EEXP; ++e) if (e != i1 && p[e] > p[i2]) i2 = e;
        const float s2 = p[i1] + p[i2];
        float* wr = wte + (size_t)t * EEXP;
#pragma unroll
        for (int e = 0; e < EEXP; ++e) wr[e] = 0.f;
        wr[i1] = p[i1] / s2;
        wr[i2] = p[i2] / s2;
    }
}

// ---------------------------------------------------------------------------
// Kernel 2: build B' [ODIM][K2] bf16 = [ W | lora_B concat over (e,r) ]
// ---------------------------------------------------------------------------
__global__ __launch_bounds__(256) void build_bprime(const float* __restrict__ W,
                                                    const float* __restrict__ loraB,
                                                    unsigned short* __restrict__ Bp) {
    const size_t idx = (size_t)blockIdx.x * 256 + threadIdx.x;   // exact grid
    const int o = (int)(idx / K2);
    const int c = (int)(idx % K2);
    float v;
    if (c < DDIM) {
        v = W[(size_t)o * DDIM + c];
    } else {
        const int e = (c - DDIM) >> 5;
        const int r = (c - DDIM) & 31;
        v = loraB[((size_t)e * ODIM + o) * RRANK + r];
    }
    Bp[idx] = f2bf(v);
}

// Kernel 3: lora_A fp32 [E*R][D] -> bf16 ([256][3072] contiguous)
__global__ __launch_bounds__(256) void convert_loraA(const float* __restrict__ loraA,
                                                     unsigned short* __restrict__ La) {
    const size_t idx = (size_t)blockIdx.x * 256 + threadIdx.x;
    La[idx] = f2bf(loraA[idx]);
}

// ---------------------------------------------------------------------------
// Shared WMMA GEMM core: C[BMxBN] += A[BMxK] * B[BNxK]^T
//   - async global->LDS (ASYNCcnt) double-buffered, 1 barrier per K-step
//   - 8 waves, 64x64 register tile per wave (16 WMMA / step)
//   LORA=false: A=A'(ld K2), B=B'(ld K2), out = fp32 [T][O]
//   LORA=true : A=A' x-part,  B=lora_A bf16 (ld D), epilogue writes
//               bf16(SCALING * w_te * acc) into A'[:, 3072+n]
// ---------------------------------------------------------------------------
template <bool LORA>
__global__ __launch_bounds__(256) void gemm_bf16(const unsigned short* __restrict__ Aq,
                                                 const unsigned short* __restrict__ Bq,
                                                 const int ldb, const int nk,
                                                 float* __restrict__ out,
                                                 unsigned short* __restrict__ outA,
                                                 const float* __restrict__ wte) {
    __shared__ unsigned short As[2][BM * LDSS];
    __shared__ unsigned short Bs[2][BN * LDSS];
    const int tid = threadIdx.x;
    const int bm0 = blockIdx.x * BM;
    const int bn0 = blockIdx.y * BN;
    const int row = tid >> 1;              // 0..127
    const int hf  = (tid & 1) << 3;        // element offset 0 or 8 -> chunks {0-7,16-23} / {8-15,24-31}

    const unsigned short* gA  = Aq + (size_t)(bm0 + row) * K2 + hf;
    const unsigned short* gB0 = Bq + (size_t)(bn0 + row) * ldb + hf;
    const unsigned short* gB1 = Bq + (size_t)(bn0 + row + 128) * ldb + hf;

    unsigned lA[2], lB0[2], lB1[2];
#pragma unroll
    for (int b = 0; b < 2; ++b) {
        lA[b]  = lds_off(&As[b][row * LDSS + hf]);
        lB0[b] = lds_off(&Bs[b][row * LDSS + hf]);
        lB1[b] = lds_off(&Bs[b][(row + 128) * LDSS + hf]);
    }

    auto issue = [&](int b, int kb) {
        const size_t ko = (size_t)kb * BK;
        async_b128(lA[b],       gA  + ko);
        async_b128(lA[b]  + 32, gA  + ko + 16);
        async_b128(lB0[b],      gB0 + ko);
        async_b128(lB0[b] + 32, gB0 + ko + 16);
        async_b128(lB1[b],      gB1 + ko);
        async_b128(lB1[b] + 32, gB1 + ko + 16);
    };

    const int wid = tid >> 5;
    const int wm = (wid >> 2) * 64;        // 0,64
    const int wn = (wid & 3) * 64;         // 0,64,128,192

    v8f acc[4][4];
    const v8f vz = {0.f,0.f,0.f,0.f,0.f,0.f,0.f,0.f};
#pragma unroll
    for (int i = 0; i < 4; ++i)
#pragma unroll
        for (int j = 0; j < 4; ++j) acc[i][j] = vz;

    issue(0, 0);
    for (int kb = 0; kb < nk; ++kb) {
        const int b = kb & 1;
        wait_async0();                      // my async writes into buffer b done
        __syncthreads();                    // everyone's done -> buffer b valid, buffer b^1 free
        if (kb + 1 < nk) {
            issue(b ^ 1, kb + 1);           // stream next slab while computing
            __builtin_prefetch(gA  + (size_t)(kb + 4) * BK, 0, 3);
            __builtin_prefetch(gB0 + (size_t)(kb + 4) * BK, 0, 3);
        }

        v16bf af[4], bfr[4];
#pragma unroll
        for (int i = 0; i < 4; ++i) af[i]  = frag_a(&As[b][(wm + i * 16) * LDSS], LDSS);
#pragma unroll
        for (int j = 0; j < 4; ++j) bfr[j] = frag_b(&Bs[b][(wn + j * 16) * LDSS], LDSS);
#pragma unroll
        for (int i = 0; i < 4; ++i)
#pragma unroll
            for (int j = 0; j < 4; ++j)
                acc[i][j] = __builtin_amdgcn_wmma_f32_16x16x32_bf16(
                    false, af[i], false, bfr[j], (short)0, acc[i][j], false, false);
    }

    const int lane = tid & 31;
    const int ln = lane & 15;
    const int lh = (lane >> 4) << 3;
#pragma unroll
    for (int i = 0; i < 4; ++i)
#pragma unroll
        for (int j = 0; j < 4; ++j) {
            const int r0 = bm0 + wm + i * 16 + lh;
            const int c0 = bn0 + wn + j * 16 + ln;
            if constexpr (!LORA) {
#pragma unroll
                for (int v = 0; v < 8; ++v)
                    out[(size_t)(r0 + v) * ODIM + c0] = acc[i][j][v];
            } else {
                const int e = c0 >> 5;       // c0 in [0,256): expert id
#pragma unroll
                for (int v = 0; v < 8; ++v) {
                    const int r = r0 + v;
                    const float wv = wte[(size_t)r * EEXP + e] * SCALING;
                    outA[(size_t)r * K2 + DDIM + c0] = f2bf(acc[i][j][v] * wv);
                }
            }
        }
}

// ---------------------------------------------------------------------------
extern "C" void kernel_launch(void* const* d_in, const int* in_sizes, int n_in,
                              void* d_out, int out_size, void* d_ws, size_t ws_size,
                              hipStream_t stream) {
    (void)in_sizes; (void)n_in; (void)out_size; (void)ws_size;
    const float* x     = (const float*)d_in[0];   // [T, D]
    const float* W     = (const float*)d_in[1];   // [O, D]
    const float* gatew = (const float*)d_in[2];   // [E, D]
    const float* loraA = (const float*)d_in[3];   // [E, R, D]
    const float* loraB = (const float*)d_in[4];   // [E, O, R]
    float* out = (float*)d_out;

    char* ws = (char*)d_ws;
    const size_t OFF_B  = (size_t)T_TOK * K2 * 2;                 // A' bytes
    const size_t OFF_LA = OFF_B + (size_t)ODIM * K2 * 2;          // B' bytes
    const size_t OFF_W  = OFF_LA + (size_t)KLORA * DDIM * 2;      // lora_A bf16 bytes
    unsigned short* Ap  = (unsigned short*)ws;
    unsigned short* Bp  = (unsigned short*)(ws + OFF_B);
    unsigned short* La  = (unsigned short*)(ws + OFF_LA);
    float*          wte = (float*)(ws + OFF_W);

    route_convert_x<<<T_TOK, 256, 0, stream>>>(x, gatew, Ap, wte);
    build_bprime<<<(unsigned)(((size_t)ODIM * K2) / 256), 256, 0, stream>>>(W, loraB, Bp);
    convert_loraA<<<(unsigned)(((size_t)KLORA * DDIM) / 256), 256, 0, stream>>>(loraA, La);
    // LoRA 'a' GEMM: N = 256 -> one macro-column
    gemm_bf16<true><<<dim3(T_TOK / BM, KLORA / BN), 256, 0, stream>>>(
        Ap, La, DDIM, DDIM / BK, nullptr, Ap, wte);
    // Main fused GEMM: out = A'[T][3328] @ B'[3072][3328]^T
    gemm_bf16<false><<<dim3(T_TOK / BM, ODIM / BN), 256, 0, stream>>>(
        Ap, Bp, K2, K2 / BK, out, nullptr, nullptr);
}
